// SGRSimilarityHead_17463337026079
// MI455X (gfx1250) — matrix operand order repaired and checked by reading, not compiled, not run on hardware
//
#include <hip/hip_runtime.h>

// ---------------------------------------------------------------------------
// SGR similarity head for MI455X (gfx1250, wave32, WMMA bf16)
// All WMMA operands live in bf16 (global pre-swizzled B tiles, bf16 LDS A
// tiles), so every inner k-step is: two 16B A loads + one 32B B load + wmma.
// ---------------------------------------------------------------------------

#define DEV static __device__ __forceinline__

typedef __bf16  v16bf __attribute__((ext_vector_type(16)));
typedef __bf16  v8bf  __attribute__((ext_vector_type(8)));
typedef float   v8f   __attribute__((ext_vector_type(8)));

static constexpr int D     = 1024;   // embedding dim
static constexpr int S     = 256;    // sim dim
static constexpr int NI    = 64;     // images
static constexpr int NC    = 64;     // captions
static constexpr int T_IMG = 196;    // image tokens (197 minus CLS)
static constexpr int T_PAD = 224;    // padded tokens (13x16 M-tiles, K=224 ctx GEMM)
static constexpr int L_CAP = 32;     // caption tokens
static constexpr int NROW  = 33;     // sim_emb rows (1 global + 32 local)
static constexpr int TILE_FL = 512;  // bf16 elems per swizzled 32x16 B tile

DEV __bf16 f2bf(float f) {
  unsigned u = __builtin_bit_cast(unsigned, f);
  unsigned r = u + 0x7FFFu + ((u >> 16) & 1u);
  unsigned short h = (unsigned short)(r >> 16);
  return __builtin_bit_cast(__bf16, h);
}
DEV float bf2f(__bf16 b) {
  unsigned u = ((unsigned)__builtin_bit_cast(unsigned short, b)) << 16;
  return __builtin_bit_cast(float, u);
}
DEV __bf16 bf_zero() { return __builtin_bit_cast(__bf16, (unsigned short)0); }
DEV float l2scale(float ss) { return 1.0f / fmaxf(sqrtf(ss), 1e-12f); }

// branch-free tanh: one v_exp + one v_rcp
DEV float fast_tanh(float x) {
  float ax = fabsf(x);
  float t = __expf(-2.0f * ax);
  float r = (1.0f - t) * __builtin_amdgcn_rcpf(1.0f + t);
  return copysignf(r, x);
}

DEV v8f wmma_bf16(v16bf a, v16bf b, v8f c) {
  return __builtin_amdgcn_wmma_f32_16x16x32_bf16(
      false, a, false, b, (short)0, c, false, false);
}

// ---- operand loaders (all unconditional, padded bf16 buffers) --------------
// A tile 16x32 from row-major bf16; per lane two contiguous 16B runs.
DEV v16bf load_a_bf(const __bf16* __restrict__ src, int ld, int m0, int k0,
                    int lane) {
  const __bf16* p = src + (size_t)(m0 + (lane & 15)) * ld + k0 + ((lane >> 4) << 4);
  v8bf lo = *(const v8bf*)p;          // K = kb .. kb+7
  v8bf hi = *(const v8bf*)(p + 8);    // K = kb+8 .. kb+15
  return __builtin_shufflevector(lo, hi, 0, 1, 2, 3, 4, 5, 6, 7,
                                 8, 9, 10, 11, 12, 13, 14, 15);
}

// B tile from pre-swizzled layout: one contiguous 32B vector per lane.
DEV v16bf load_b_sw(const __bf16* __restrict__ buf, int ntiles, int k0, int n0,
                    int lane) {
  return *(const v16bf*)(buf +
      ((size_t)(k0 >> 5) * ntiles + (n0 >> 4)) * TILE_FL + lane * 16);
}

// B tile where logical B[k][n] = src[n*ld + k] (K contiguous in memory).
DEV v16bf load_bt_nc(const __bf16* __restrict__ src, int ld, int k0, int n0,
                     int lane) {
  return *(const v16bf*)(src + (size_t)(n0 + (lane & 15)) * ld + k0 +
                         ((lane >> 4) << 4));
}

// ---------------------------------------------------------------------------
// Kernel: fp32 -> bf16 flat conversion (W_glo keeps plain layout)
// ---------------------------------------------------------------------------
__global__ void k_cvt_bf16(const float* __restrict__ src,
                           __bf16* __restrict__ dst, int n) {
  for (int i = blockIdx.x * blockDim.x + threadIdx.x; i < n;
       i += gridDim.x * blockDim.x)
    dst[i] = f2bf(src[i]);
}

// ---------------------------------------------------------------------------
// Kernel: swizzle a KxN fp32 matrix into WMMA B-tile order (bf16)
// ---------------------------------------------------------------------------
__global__ void k_swz_b(const float* __restrict__ src, __bf16* __restrict__ dst,
                        int K, int N) {
  int ntiles = N >> 4;
  int total = K * N;
  for (int idx = blockIdx.x * blockDim.x + threadIdx.x; idx < total;
       idx += gridDim.x * blockDim.x) {
    int e = idx & 15, lane = (idx >> 4) & 31, t = idx >> 9;
    int nt = t % ntiles, kt = t / ntiles;
    int k = (kt << 5) + ((lane >> 4) << 4) + e;
    int n = (nt << 4) + (lane & 15);
    dst[idx] = f2bf(src[(size_t)k * N + n]);
  }
}

// Batched: image tokens (Tvalid x 1024, zero pad to Kp rows) into B-tile order.
__global__ void k_swz_tok(const float* __restrict__ src, size_t stride,
                          int Tvalid, int Kp, __bf16* __restrict__ dst) {
  const float* s = src + (size_t)blockIdx.y * stride;
  int ktiles = Kp >> 5;
  size_t per = (size_t)ktiles * (D >> 4) * TILE_FL;
  __bf16* d = dst + (size_t)blockIdx.y * per;
  int total = (int)per;
  for (int idx = blockIdx.x * blockDim.x + threadIdx.x; idx < total;
       idx += gridDim.x * blockDim.x) {
    int e = idx & 15, lane = (idx >> 4) & 31, t = idx >> 9;
    int nt = t & 63, kt = t >> 6;
    int k = (kt << 5) + ((lane >> 4) << 4) + e;
    int n = (nt << 4) + (lane & 15);
    float v = (k < Tvalid) ? s[(size_t)k * D + n] : 0.0f;
    d[idx] = f2bf(v);
  }
}

// ---------------------------------------------------------------------------
// Kernel: token prep — bf16 copy (zero-padded rows) + bf16 mean over tokens
// ---------------------------------------------------------------------------
__global__ void k_tok_prep(const float* __restrict__ src, size_t src_stride,
                           int T, int Tpad, __bf16* __restrict__ dst_bf,
                           __bf16* __restrict__ avg_bf) {
  int b = blockIdx.x;
  const float* tok = src + (size_t)b * src_stride;
  __bf16* dst = dst_bf + (size_t)b * Tpad * D;
  for (int e = threadIdx.x; e < Tpad * D; e += blockDim.x) {
    int t = e >> 10;
    dst[e] = f2bf((t < T) ? tok[e] : 0.0f);
  }
  float inv = 1.0f / (float)T;
  for (int d = threadIdx.x; d < D; d += blockDim.x) {
    float s = 0.0f;
    for (int t = 0; t < T; ++t) s += tok[t * D + d];
    avg_bf[b * D + d] = f2bf(s * inv);
  }
}

// ---------------------------------------------------------------------------
// Kernel: C = tanh(A(MxK,bf16) @ Bsw(KxN) + bias), one wave per 16x16 tile
// ---------------------------------------------------------------------------
__global__ void k_gemm_tanh(const __bf16* __restrict__ A,
                            const __bf16* __restrict__ Bsw,
                            const float* __restrict__ bias,
                            float* __restrict__ C, int M, int N, int K) {
  int lane = threadIdx.x & 31;
  int m0 = blockIdx.x * 16, n0 = blockIdx.y * 16;
  int ntiles = N >> 4;
  v8f acc = {};
  for (int k0 = 0; k0 < K; k0 += 32) {
    v16bf a = load_a_bf(A, K, m0, k0, lane);
    v16bf b = load_b_sw(Bsw, ntiles, k0, n0, lane);
    acc = wmma_bf16(a, b, acc);
  }
  int col = n0 + (lane & 15);
  int roff = (lane >> 4) << 3;
#pragma unroll
  for (int e = 0; e < 8; ++e) {
    int row = m0 + roff + e;
    if (row < M) C[(size_t)row * N + col] = fast_tanh(acc[e] + bias[col]);
  }
}

// ---------------------------------------------------------------------------
// Kernel: global self-attention pooling (one block per image/caption).
// ---------------------------------------------------------------------------
__global__ void k_global_attn(const __bf16* __restrict__ tok_bf, size_t sbf,
                              const float* __restrict__ tok_f, size_t sf,
                              int T, const __bf16* __restrict__ Wl_sw,
                              const float* __restrict__ bl,
                              const float* __restrict__ GE,
                              const float* __restrict__ Wc,
                              const float* __restrict__ bc,
                              float* __restrict__ outG) {
  __shared__ float gw[D];
  __shared__ float logits[208];
  __shared__ float gf[D];
  __shared__ float red[1];
  int b = blockIdx.x;
  const __bf16* tb = tok_bf + (size_t)b * sbf;
  const float*  tf = tok_f  + (size_t)b * sf;
  int lane = threadIdx.x & 31, wave = threadIdx.x >> 5;
  const int NW = blockDim.x >> 5;

  for (int d = threadIdx.x; d < D; d += blockDim.x) gw[d] = GE[b * D + d] * Wc[d];
  for (int t = threadIdx.x; t < 208; t += blockDim.x) logits[t] = 0.0f;
  __syncthreads();

  int mt = (T + 15) >> 4;
  for (int tile = wave; tile < mt * (D / 16); tile += NW) {
    int m0 = (tile / (D / 16)) * 16, n0 = (tile % (D / 16)) * 16;
    v8f acc = {};
    for (int k0 = 0; k0 < D; k0 += 32) {
      __builtin_prefetch(Wl_sw + ((size_t)((k0 + 32) >> 5) * 64 + (n0 >> 4)) * TILE_FL, 0, 1);
      v16bf a  = load_a_bf(tb, D, m0, k0, lane);
      v16bf bm = load_b_sw(Wl_sw, 64, k0, n0, lane);
      acc = wmma_bf16(a, bm, acc);
    }
    int col = n0 + (lane & 15);
    int roff = (lane >> 4) << 3;
#pragma unroll
    for (int e = 0; e < 8; ++e) {
      int row = m0 + roff + e;
      if (row < T) {
        float v = fast_tanh(acc[e] + bl[col]);
        atomicAdd(&logits[row], v * gw[col]);
      }
    }
  }
  __syncthreads();

  if (threadIdx.x == 0) {
    float c0 = bc[0];
    float mx = -1e30f;
    for (int t = 0; t < T; ++t) mx = fmaxf(mx, logits[t] + c0);
    float s = 0.0f;
    for (int t = 0; t < T; ++t) { float e = __expf(logits[t] + c0 - mx); logits[t] = e; s += e; }
    float inv = 1.0f / s;
    for (int t = 0; t < T; ++t) logits[t] *= inv;
    red[0] = 0.0f;
  }
  __syncthreads();

  for (int d = threadIdx.x; d < D; d += blockDim.x) {
    float s = 0.0f;
    for (int t = 0; t < T; ++t) s += logits[t] * tf[(size_t)t * D + d];
    gf[d] = s;
    atomicAdd(&red[0], s * s);
  }
  __syncthreads();
  float sc = l2scale(red[0]);
  for (int d = threadIdx.x; d < D; d += blockDim.x) outG[(size_t)b * D + d] = gf[d] * sc;
}

// ---------------------------------------------------------------------------
// Kernel: fused per-pair pipeline (one block per (img, cap) pair)
// LDS arena (floats): attn | wq(bf16) | Breg(ctx_bf/Q,K) | X(bf16) | Y(bf16)
//                     | edge | dg | red    — ~167 KB of the 320 KB WGP LDS
// ---------------------------------------------------------------------------
static constexpr int ATTN_FL = 208 * 32;          // fp32 attention scores
static constexpr int WQB_FL  = (32 * 224) / 2;    // bf16 softmax weights
static constexpr int BREG_FL = 2 * NROW * S;      // ctx_bf16 or Q,K (fp32)
static constexpr int XB_FL   = (48 * S) / 2;      // bf16 sim_emb, padded rows
static constexpr int YB_FL   = (48 * S) / 2;      // bf16 edge@X, padded rows
static constexpr int EDGE_FL = 1120;
static constexpr int DG_FL   = D;
static constexpr int RED_FL  = 64;
static constexpr int PAIR_SMEM_FL =
    ATTN_FL + WQB_FL + BREG_FL + XB_FL + YB_FL + EDGE_FL + DG_FL + RED_FL;

__global__ void k_pair(const __bf16* __restrict__ imgT_bf,   // [NI][224][1024]
                       const __bf16* __restrict__ imgB_sw,   // B-tile layout
                       const __bf16* __restrict__ cap_bf,    // [NC][32][1024]
                       const float* __restrict__ cap_f,
                       const float* __restrict__ img_global,
                       const float* __restrict__ cap_global,
                       const __bf16* __restrict__ Wloc_sw, const float* __restrict__ bloc,
                       const __bf16* __restrict__ Wglo_bf, const float* __restrict__ bglo,
                       const __bf16* __restrict__ sWq, const float* __restrict__ sbq,
                       const __bf16* __restrict__ sWk, const float* __restrict__ sbk,
                       const __bf16* __restrict__ sWg, const float* __restrict__ sbg,
                       const float* __restrict__ W_eval, const float* __restrict__ b_eval,
                       float* __restrict__ out) {
  extern __shared__ float sm[];
  float*  attn = sm;                          // [208][32] fp32
  __bf16* wqb  = (__bf16*)(attn + ATTN_FL);   // [32][224] bf16
  float*  Breg = attn + ATTN_FL + WQB_FL;
  __bf16* Xb   = (__bf16*)(Breg + BREG_FL);   // [48][256] bf16, pad rows = 0
  __bf16* Yb   = (__bf16*)(Breg + BREG_FL + XB_FL);  // [48][256] bf16
  float*  edge = Breg + BREG_FL + XB_FL + YB_FL;     // [33][33]
  float*  dg   = edge + EDGE_FL;              // [1024]
  float*  red  = dg + DG_FL;                  // [64]
  __bf16* ctxb = (__bf16*)Breg;               // [32][1024] bf16 view

  int i = blockIdx.x, c = blockIdx.y;
  int lane = threadIdx.x & 31, wave = threadIdx.x >> 5;
  const int NW = blockDim.x >> 5;
  const __bf16* tb = imgT_bf + (size_t)i * T_PAD * D;
  const __bf16* tB = imgB_sw + (size_t)i * (T_PAD >> 5) * (D >> 4) * TILE_FL;
  const __bf16* cb = cap_bf  + (size_t)c * L_CAP * D;
  const float*  cf = cap_f   + (size_t)c * L_CAP * D;

  // ---- attn = leaky_relu( tokens(196x1024) @ cap^T ) -----------------------
  for (int tile = wave; tile < 13 * 2; tile += NW) {
    int m0 = (tile >> 1) * 16, n0 = (tile & 1) * 16;
    v8f acc = {};
    for (int k0 = 0; k0 < D; k0 += 32) {
      v16bf a  = load_a_bf(tb, D, m0, k0, lane);
      v16bf bm = load_bt_nc(cb, D, k0, n0, lane);
      acc = wmma_bf16(a, bm, acc);
    }
    int col = n0 + (lane & 15);
    int roff = (lane >> 4) << 3;
#pragma unroll
    for (int e = 0; e < 8; ++e) {
      int row = m0 + roff + e;
      float v = acc[e];
      v = (v > 0.0f) ? v : 0.1f * v;
      attn[row * 32 + col] = (row < T_IMG) ? v : 0.0f;
    }
  }
  __syncthreads();

  for (int row = threadIdx.x; row < T_IMG; row += blockDim.x) {
    float ss = 0.0f;
    for (int q = 0; q < 32; ++q) { float v = attn[row * 32 + q]; ss += v * v; }
    float sc = l2scale(ss);
    for (int q = 0; q < 32; ++q) attn[row * 32 + q] *= sc;
  }
  __syncthreads();

  // softmax over context per query (smooth=9); write bf16 weights, K-pad 224
  for (int q = threadIdx.x; q < 32; q += blockDim.x) {
    float mx = -1e30f;
    for (int t = 0; t < T_IMG; ++t) mx = fmaxf(mx, attn[t * 32 + q]);
    float s = 0.0f;
    for (int t = 0; t < T_IMG; ++t) s += __expf(9.0f * (attn[t * 32 + q] - mx));
    float inv = 1.0f / s;
    for (int t = 0; t < T_IMG; ++t)
      wqb[q * 224 + t] = f2bf(__expf(9.0f * (attn[t * 32 + q] - mx)) * inv);
    for (int t = T_IMG; t < 224; ++t) wqb[q * 224 + t] = bf_zero();
  }
  for (int r = threadIdx.x; r < RED_FL; r += blockDim.x) red[r] = 0.0f;
  __syncthreads();

  // ---- ctx = wq(32x224) @ tokens(224x1024, B-swizzled) ---------------------
  for (int tile = wave; tile < 2 * 64; tile += NW) {
    int m0 = (tile >> 6) * 16, n0 = (tile & 63) * 16;
    v8f acc = {};
    for (int k0 = 0; k0 < 224; k0 += 32) {
      v16bf a  = load_a_bf(wqb, 224, m0, k0, lane);
      v16bf bm = load_b_sw(tB, 64, k0, n0, lane);
      acc = wmma_bf16(a, bm, acc);
    }
    int col = n0 + (lane & 15);
    int roff = (lane >> 4) << 3;
#pragma unroll
    for (int e = 0; e < 8; ++e) {
      int row = m0 + roff + e;
      float v = acc[e];
      ctxb[row * D + col] = f2bf(v);
      atomicAdd(&red[row], v * v);
    }
  }
  __syncthreads();

  // diff2 = (l2norm(ctx) - cap)^2 in place (bf16)
  for (int e = threadIdx.x; e < L_CAP * D; e += blockDim.x) {
    int row = e >> 10;
    float sc = l2scale(red[row]);
    float v = bf2f(ctxb[e]) * sc - cf[e];
    ctxb[e] = f2bf(v * v);
  }
  __syncthreads();

  for (int e = threadIdx.x; e < 48 * S; e += blockDim.x) Xb[e] = bf_zero();
  for (int r = threadIdx.x; r < RED_FL; r += blockDim.x) red[r] = 0.0f;
  __syncthreads();

  // ---- sim_loc: X rows 1..32 = diff2(32x1024) @ Wloc + bloc ---------------
  for (int tile = wave; tile < 2 * 16; tile += NW) {
    int m0 = (tile >> 4) * 16, n0 = (tile & 15) * 16;
    v8f acc = {};
    for (int k0 = 0; k0 < D; k0 += 32) {
      __builtin_prefetch(Wloc_sw + ((size_t)((k0 + 32) >> 5) * 16 + (n0 >> 4)) * TILE_FL, 0, 1);
      v16bf a  = load_a_bf(ctxb, D, m0, k0, lane);
      v16bf bm = load_b_sw(Wloc_sw, 16, k0, n0, lane);
      acc = wmma_bf16(a, bm, acc);
    }
    int col = n0 + (lane & 15);
    int roff = (lane >> 4) << 3;
#pragma unroll
    for (int e = 0; e < 8; ++e) {
      int row = m0 + roff + e;
      float v = acc[e] + bloc[col];
      Xb[(row + 1) * S + col] = f2bf(v);
      atomicAdd(&red[row + 1], v * v);
    }
  }
  // ---- sim_glo: X row 0 ----------------------------------------------------
  const float* ig = img_global + (size_t)i * D;
  const float* cg = cap_global + (size_t)c * D;
  for (int d = threadIdx.x; d < D; d += blockDim.x) {
    float v = ig[d] - cg[d];
    dg[d] = v * v;
  }
  __syncthreads();
  for (int n = threadIdx.x; n < S; n += blockDim.x) {
    float s = bglo[n];
    for (int k = 0; k < D; ++k) s += dg[k] * bf2f(Wglo_bf[(size_t)k * S + n]);
    Xb[n] = f2bf(s);
    atomicAdd(&red[0], s * s);
  }
  __syncthreads();
  for (int e = threadIdx.x; e < NROW * S; e += blockDim.x) {
    int row = e >> 8;
    Xb[e] = f2bf(bf2f(Xb[e]) * l2scale(red[row]));
  }
  __syncthreads();

  // ---- 3x SGR graph reasoning ---------------------------------------------
  for (int step = 0; step < 3; ++step) {
    const __bf16* Wq = sWq + (size_t)step * S * S;
    const __bf16* Wk = sWk + (size_t)step * S * S;
    const __bf16* Wg = sWg + (size_t)step * S * S;
    const float* bq = sbq + step * S;
    const float* bk = sbk + step * S;
    const float* bg = sbg + step * S;
    float* Q  = Breg;
    float* Kb = Breg + NROW * S;

    for (int tile = wave; tile < 3 * 16 * 2; tile += NW) {
      int mat = tile & 1;
      int t2 = tile >> 1;
      int m0 = (t2 >> 4) * 16, n0 = (t2 & 15) * 16;
      const __bf16* W = mat ? Wk : Wq;
      const float* bb = mat ? bk : bq;
      float* Dst = mat ? Kb : Q;
      v8f acc = {};
      for (int k0 = 0; k0 < S; k0 += 32) {
        v16bf a  = load_a_bf(Xb, S, m0, k0, lane);
        v16bf bm = load_b_sw(W, 16, k0, n0, lane);
        acc = wmma_bf16(a, bm, acc);
      }
      int col = n0 + (lane & 15);
      int roff = (lane >> 4) << 3;
#pragma unroll
      for (int e = 0; e < 8; ++e) {
        int row = m0 + roff + e;
        if (row < NROW) Dst[row * S + col] = acc[e] + bb[col];
      }
    }
    __syncthreads();

    for (int nm = threadIdx.x; nm < NROW * NROW; nm += blockDim.x) {
      int n = nm / NROW, m = nm % NROW;
      float s = 0.0f;
      for (int k = 0; k < S; ++k) s += Q[n * S + k] * Kb[m * S + k];
      edge[n * NROW + m] = s;
    }
    __syncthreads();
    for (int n = threadIdx.x; n < NROW; n += blockDim.x) {
      float mx = -1e30f;
      for (int m = 0; m < NROW; ++m) mx = fmaxf(mx, edge[n * NROW + m]);
      float s = 0.0f;
      for (int m = 0; m < NROW; ++m) {
        float e = __expf(edge[n * NROW + m] - mx);
        edge[n * NROW + m] = e; s += e;
      }
      float inv = 1.0f / s;
      for (int m = 0; m < NROW; ++m) edge[n * NROW + m] *= inv;
    }
    __syncthreads();

    // Y = edge @ X (48 rows, pad rows zeroed -> clean WMMA A operand)
    for (int e = threadIdx.x; e < 48 * S; e += blockDim.x) {
      int n = e >> 8, d = e & 255;
      float s = 0.0f;
      if (n < NROW)
        for (int m = 0; m < NROW; ++m) s += edge[n * NROW + m] * bf2f(Xb[m * S + d]);
      Yb[e] = f2bf(s);
    }
    __syncthreads();

    for (int tile = wave; tile < 3 * 16; tile += NW) {
      int m0 = (tile >> 4) * 16, n0 = (tile & 15) * 16;
      v8f acc = {};
      for (int k0 = 0; k0 < S; k0 += 32) {
        v16bf a  = load_a_bf(Yb, S, m0, k0, lane);
        v16bf bm = load_b_sw(Wg, 16, k0, n0, lane);
        acc = wmma_bf16(a, bm, acc);
      }
      int col = n0 + (lane & 15);
      int roff = (lane >> 4) << 3;
#pragma unroll
      for (int e = 0; e < 8; ++e) {
        int row = m0 + roff + e;
        if (row < NROW) Xb[row * S + col] = f2bf(fmaxf(acc[e] + bg[col], 0.0f));
      }
    }
    __syncthreads();
  }

  // ---- eval ---------------------------------------------------------------
  if (wave == 0) {
    float s = 0.0f;
    for (int k = lane; k < S; k += 32) s += bf2f(Xb[k]) * W_eval[k];
#pragma unroll
    for (int off = 16; off > 0; off >>= 1) s += __shfl_down(s, off, 32);
    if (lane == 0)
      out[(size_t)i * NC + c] = 1.0f / (1.0f + __expf(-(s + b_eval[0])));
  }
}

// ---------------------------------------------------------------------------
// Host launcher
// ---------------------------------------------------------------------------
extern "C" void kernel_launch(void* const* d_in, const int* in_sizes, int n_in,
                              void* d_out, int out_size, void* d_ws, size_t ws_size,
                              hipStream_t stream) {
  const float* img_embs = (const float*)d_in[0];
  const float* cap_embs = (const float*)d_in[1];
  const float* vG_Wl = (const float*)d_in[3];
  const float* vG_bl = (const float*)d_in[4];
  const float* vG_Wg = (const float*)d_in[5];
  const float* vG_bg = (const float*)d_in[6];
  const float* vG_Wc = (const float*)d_in[7];
  const float* vG_bc = (const float*)d_in[8];
  const float* tG_Wl = (const float*)d_in[9];
  const float* tG_bl = (const float*)d_in[10];
  const float* tG_Wg = (const float*)d_in[11];
  const float* tG_bg = (const float*)d_in[12];
  const float* tG_Wc = (const float*)d_in[13];
  const float* tG_bc = (const float*)d_in[14];
  const float* W_loc = (const float*)d_in[15];
  const float* b_loc = (const float*)d_in[16];
  const float* W_glo = (const float*)d_in[17];
  const float* b_glo = (const float*)d_in[18];
  const float* W_eval = (const float*)d_in[19];
  const float* b_eval = (const float*)d_in[20];
  const float* sgr_Wq = (const float*)d_in[21];
  const float* sgr_bq = (const float*)d_in[22];
  const float* sgr_Wk = (const float*)d_in[23];
  const float* sgr_bk = (const float*)d_in[24];
  const float* sgr_Wg = (const float*)d_in[25];
  const float* sgr_bg = (const float*)d_in[26];

  char* ws = (char*)d_ws;
  size_t off = 0;
  auto carve = [&](size_t bytes) -> char* {
    char* p = ws + off;
    off += (bytes + 255) & ~(size_t)255;
    return p;
  };
  __bf16* imgT_bf = (__bf16*)carve((size_t)NI * T_PAD * D * 2);
  __bf16* imgB_sw = (__bf16*)carve((size_t)NI * (T_PAD >> 5) * (D >> 4) * TILE_FL * 2);
  __bf16* cap_bf  = (__bf16*)carve((size_t)NC * L_CAP * D * 2);
  __bf16* img_avg = (__bf16*)carve((size_t)NI * D * 2);
  __bf16* cap_avg = (__bf16*)carve((size_t)NC * D * 2);
  float* GEi      = (float*)carve((size_t)NI * D * 4);
  float* GEc      = (float*)carve((size_t)NC * D * 4);
  float* img_glo  = (float*)carve((size_t)NI * D * 4);
  float* cap_glo  = (float*)carve((size_t)NC * D * 4);
  __bf16* Wl_v = (__bf16*)carve((size_t)D * D * 2);
  __bf16* Wg_v = (__bf16*)carve((size_t)D * D * 2);
  __bf16* Wl_t = (__bf16*)carve((size_t)D * D * 2);
  __bf16* Wg_t = (__bf16*)carve((size_t)D * D * 2);
  __bf16* Wloc = (__bf16*)carve((size_t)D * S * 2);
  __bf16* Wglo = (__bf16*)carve((size_t)D * S * 2);
  __bf16* sWq  = (__bf16*)carve((size_t)3 * S * S * 2);
  __bf16* sWk  = (__bf16*)carve((size_t)3 * S * S * 2);
  __bf16* sWg  = (__bf16*)carve((size_t)3 * S * S * 2);

  auto swz = [&](const float* src, __bf16* dst, int K, int N) {
    int total = K * N;
    int blocks = (total + 255) / 256;
    if (blocks > 4096) blocks = 4096;
    k_swz_b<<<blocks, 256, 0, stream>>>(src, dst, K, N);
  };
  swz(vG_Wl, Wl_v, D, D);
  swz(vG_Wg, Wg_v, D, D);
  swz(tG_Wl, Wl_t, D, D);
  swz(tG_Wg, Wg_t, D, D);
  swz(W_loc, Wloc, D, S);
  for (int t = 0; t < 3; ++t) {
    swz(sgr_Wq + (size_t)t * S * S, sWq + (size_t)t * S * S, S, S);
    swz(sgr_Wk + (size_t)t * S * S, sWk + (size_t)t * S * S, S, S);
    swz(sgr_Wg + (size_t)t * S * S, sWg + (size_t)t * S * S, S, S);
  }
  {
    int n = D * S;
    k_cvt_bf16<<<(n + 255) / 256, 256, 0, stream>>>(W_glo, Wglo, n);
  }

  // token prep (images skip CLS token; rows padded to T_PAD)
  k_tok_prep<<<NI, 256, 0, stream>>>(img_embs + D, (size_t)197 * D, T_IMG, T_PAD,
                                     imgT_bf, img_avg);
  k_tok_prep<<<NC, 256, 0, stream>>>(cap_embs, (size_t)L_CAP * D, L_CAP, L_CAP,
                                     cap_bf, cap_avg);
  // image tokens in B-tile order (K rows = tokens, zero-padded to 224)
  k_swz_tok<<<dim3(1024, NI), 256, 0, stream>>>(img_embs + D, (size_t)197 * D,
                                                T_IMG, T_PAD, imgB_sw);

  // ge = tanh(avg @ Wg + bg)
  k_gemm_tanh<<<dim3(NI / 16, D / 16), 32, 0, stream>>>(img_avg, Wg_v, vG_bg,
                                                        GEi, NI, D, D);
  k_gemm_tanh<<<dim3(NC / 16, D / 16), 32, 0, stream>>>(cap_avg, Wg_t, tG_bg,
                                                        GEc, NC, D, D);

  // global self-attention pooled features
  k_global_attn<<<NI, 256, 0, stream>>>(imgT_bf, (size_t)T_PAD * D,
                                        img_embs + D, (size_t)197 * D, T_IMG,
                                        Wl_v, vG_bl, GEi, vG_Wc, vG_bc, img_glo);
  k_global_attn<<<NC, 256, 0, stream>>>(cap_bf, (size_t)L_CAP * D,
                                        cap_embs, (size_t)L_CAP * D, L_CAP,
                                        Wl_t, tG_bl, GEc, tG_Wc, tG_bc, cap_glo);

  // fused per-pair pipeline
  size_t pair_smem = (size_t)PAIR_SMEM_FL * sizeof(float);   // ~167 KB / WGP
  k_pair<<<dim3(NI, NC), 256, pair_smem, stream>>>(
      imgT_bf, imgB_sw, cap_bf, cap_embs, img_glo, cap_glo,
      Wloc, b_loc, Wglo, b_glo,
      sWq, sgr_bq, sWk, sgr_bk, sWg, sgr_bg,
      W_eval, b_eval, (float*)d_out);

  (void)in_sizes; (void)n_in; (void)out_size; (void)ws_size;
}
